// SwinCIR_52080773431476
// MI455X (gfx1250) — matrix operand order repaired and checked by reading, not compiled, not run on hardware
//
#include <hip/hip_runtime.h>
#include <stdint.h>

typedef __attribute__((ext_vector_type(16))) __bf16   v16bf;
typedef __attribute__((ext_vector_type(8)))  float    v8f;
typedef __attribute__((ext_vector_type(4)))  uint32_t u32x4;
typedef __attribute__((ext_vector_type(8)))  uint32_t u32x8;

#define WSZ   8
#define NTOK  64
#define CCH   60
#define CP    64
#define NH    6
#define DH    10
#define DP    32
#define QKVC  180
#define QKVP  192
#define HIMG  384
#define WIMG  384
#define NWH   48
#define NWW   48
#define WNUM  (NWH * NWW)
#define BIMG  4
#define NBIAS (15 * 15 * NH)      // 1350 floats

// staging blob layout (bytes): [0,24576) WqkvT bf16 [192][64] (Q cols pre-scaled by 1/sqrt(d))
//                              [24576,32768) WpT bf16 [64][64]
//                              [32768,38168) bias f32 [1350]
#define BLOB_WQ   0
#define BLOB_WP   24576
#define BLOB_BS   32768
#define BLOB_SZ   38176           // padded to 16B
#define BLOB_NE   (BLOB_SZ / 4)   // 9544 4-byte elements

// float -> bf16 (round to nearest even), raw u16
__device__ __forceinline__ uint16_t f2bf(float f) {
    uint32_t u = __builtin_bit_cast(uint32_t, f);
    uint32_t r = u + 0x7FFFu + ((u >> 16) & 1u);
    return (uint16_t)(r >> 16);
}

union FragU {
    uint4 q[2];
    v16bf v;
};

// A-fragment (16x32 bf16). lane L<16: row M=L, K = {0..7,16..23}; lane L+16: K = {8..15,24..31}.
__device__ __forceinline__ v16bf loadA(const uint16_t* base, int stride, int lane) {
    int M  = lane & 15;
    int kh = (lane >> 4) << 3;
    FragU u;
    const uint16_t* p = base + M * stride + kh;
    u.q[0] = *(const uint4*)(p);
    u.q[1] = *(const uint4*)(p + 16);
    return u.v;
}

// B-fragment (32x16 bf16) from transposed store BT[n][k] (contiguous in K).
__device__ __forceinline__ v16bf loadBT(const uint16_t* base, int stride, int lane) {
    int N  = lane & 15;
    int kg = (lane >> 4) << 3;
    FragU u;
    const uint16_t* p = base + N * stride + kg;
    u.q[0] = *(const uint4*)(p);
    u.q[1] = *(const uint4*)(p + 16);
    return u.v;
}

// ---- prep: bake transposed bf16 weights (+1/sqrt(d) folded into Q cols) + bias blob ----
__global__ void swin_prep(const float* __restrict__ qkv_w,
                          const float* __restrict__ proj_w,
                          const float* __restrict__ bias_m,
                          uint32_t* __restrict__ blob)
{
    const int   tid    = threadIdx.x;
    const float qscale = 0.31622776601683794f; // 1/sqrt(10)
    uint16_t* wq = (uint16_t*)((uint8_t*)blob + BLOB_WQ);
    uint16_t* wp = (uint16_t*)((uint8_t*)blob + BLOB_WP);
    float*    bs = (float*)((uint8_t*)blob + BLOB_BS);

    for (int i = tid; i < QKVP * CP; i += 256) {
        int c = i >> 6, k = i & 63;
        uint16_t v = 0;
        if (c < QKVC && k < CCH) {
            float w = qkv_w[k * QKVC + c];
            if (c < CCH) w *= qscale;          // Q block pre-scaled
            v = f2bf(w);
        }
        wq[i] = v;
    }
    for (int i = tid; i < CP * CP; i += 256) {
        int c = i >> 6, k = i & 63;
        uint16_t v = 0;
        if (c < CCH && k < CCH) v = f2bf(proj_w[k * CCH + c]);
        wp[i] = v;
    }
    for (int i = tid; i < NBIAS; i += 256) bs[i] = bias_m[i];
    if (tid == 0) { ((uint32_t*)((uint8_t*)blob + 38168))[0] = 0;
                    ((uint32_t*)((uint8_t*)blob + 38168))[1] = 0; }
}

__global__ __launch_bounds__(256, 1)
void swin_wmsa_fused(const float* __restrict__ img,
                     const uint32_t* __restrict__ blob,
                     const float* __restrict__ proj_b,
                     float* __restrict__ out)
{
    __shared__ __align__(16) uint8_t  Stage[BLOB_SZ];        // 37.3 KB (TDM target)
    __shared__ __align__(16) uint16_t Xs[NTOK][CP];          //  8 KB
    __shared__ __align__(16) uint16_t QKs[2][NH][NTOK][DP];  // 48 KB (0=Q pre-scaled, 1=K)
    __shared__ __align__(16) uint16_t VTs[NH][16][NTOK];     // 12 KB (V transposed [d][m])
    __shared__ __align__(16) float    Ss[2][NTOK][NTOK];     // 32 KB
    __shared__ __align__(16) uint16_t Ps[2][NTOK][NTOK];     // 16 KB
    __shared__ __align__(16) uint16_t Os[NTOK][CP];          //  8 KB

    const uint16_t (*WqkvT)[CP] = (const uint16_t(*)[CP])&Stage[BLOB_WQ];
    const uint16_t (*WpT)[CP]   = (const uint16_t(*)[CP])&Stage[BLOB_WP];
    const float*   BsL          = (const float*)&Stage[BLOB_BS];

    const int tid  = threadIdx.x;
    const int wid  = tid >> 5;
    const int lane = tid & 31;

    const int widx = blockIdx.x;
    const int b    = widx / WNUM;
    const int wrem = widx % WNUM;
    const int wh   = wrem / NWW;
    const int ww   = wrem % NWW;

    // ---- TDM: DMA the 37 KB weight/bias blob into LDS (one wave issues) ----
    if (wid == 0) {
        uint32_t lds_base = (uint32_t)(uintptr_t)(&Stage[0]);
        uint64_t ga       = (uint64_t)(uintptr_t)blob;
        u32x4 g0;
        g0[0] = 1u;                                  // count=1 (valid descriptor)
        g0[1] = lds_base;                            // lds_addr
        g0[2] = (uint32_t)ga;                        // global_addr[31:0]
        g0[3] = (uint32_t)(ga >> 32) | 0x80000000u;  // global_addr[56:32] | type=2
        u32x8 g1;
        g1[0] = 2u << 16;                            // data_size = 4 bytes
        g1[1] = (BLOB_NE & 0xFFFFu) << 16;           // tensor_dim0[15:0]
        g1[2] = (BLOB_NE >> 16) | (1u << 16);        // tensor_dim0[31:16] | tensor_dim1=1
        g1[3] = (BLOB_NE & 0xFFFFu) << 16;           // tile_dim0
        g1[4] = 1u;                                  // tile_dim1=1, tile_dim2=0
        g1[5] = BLOB_NE;                             // tensor_dim0_stride[31:0]
        g1[6] = 0u;
        g1[7] = 0u;
        asm volatile("tensor_load_to_lds %0, %1" :: "s"(g0), "s"(g1) : "memory");
        __builtin_amdgcn_s_wait_tensorcnt(0);
    }

    // ---- zero pad regions (overlaps with TDM) ----
    for (int i = tid; i < 2 * NH * NTOK * DP; i += 256) (&QKs[0][0][0][0])[i] = 0;
    for (int i = tid; i < NH * 16 * NTOK; i += 256)     (&VTs[0][0][0])[i]    = 0;
    for (int i = tid; i < NTOK * CP; i += 256)          (&Os[0][0])[i]        = 0;

    // ---- stage shifted window X -> bf16 (roll -w/2: src = pos + 4 mod 384) ----
    for (int i = tid; i < NTOK * CP; i += 256) {
        int n = i >> 6, k = i & 63;
        int ii = n >> 3, jj = n & 7;
        int hs = (wh * WSZ + ii + WSZ / 2) % HIMG;
        int ws = (ww * WSZ + jj + WSZ / 2) % WIMG;
        uint16_t v = 0;
        if (k < CCH)
            v = f2bf(img[(((size_t)b * HIMG + hs) * WIMG + ws) * CCH + k]);
        Xs[n][k] = v;
    }
    __syncthreads();

    // ======== GEMM1: QKV[64 x 192] = Xs[64 x 64] * Wqkv ========
    for (int t = wid; t < 48; t += 8) {
        int mt = t & 3, nt = t >> 2;               // shifts only
        v8f acc = {};
#pragma unroll
        for (int kt = 0; kt < 2; ++kt) {
            v16bf a  = loadA(&Xs[mt * 16][0] + kt * 32, CP, lane);
            v16bf bb = loadBT(&WqkvT[nt * 16][0] + kt * 32, CP, lane);
            acc = __builtin_amdgcn_wmma_f32_16x16x32_bf16(false, a, false, bb,
                                                          (short)0, acc, false, false);
        }
        int n0 = mt * 16 + ((lane >> 4) << 3);
        int c  = nt * 16 + (lane & 15);
        if (c < QKVC) {
            int e   = (c >= 120) ? 2 : (c >= 60 ? 1 : 0);
            int rem = c - e * 60;
            int h   = (rem * 205) >> 11;           // rem / 10
            int dd  = rem - h * 10;
            // single-select scatter: Q/K share layout; V is the transposed store
            uint16_t* dstQK = &QKs[e & 1][h][n0][dd];
            uint16_t* dstV  = &VTs[h][dd][n0];
            bool isV        = (e == 2);
            uint16_t* dst   = isV ? dstV : dstQK;
            int strideTok   = isV ? 1 : DP;
#pragma unroll
            for (int r = 0; r < 8; ++r)
                dst[r * strideTok] = f2bf(acc[r]);
        }
    }
    __syncthreads();

    const bool edgeH = (wh == NWH - 1);
    const bool edgeW = (ww == NWW - 1);

    // ======== attention: two heads per iteration ========
    for (int h0 = 0; h0 < NH; h0 += 2) {
        // S = Q * K^T (+ bias + shift mask): 32 tiles, 4 per wave
        for (int t = wid; t < 32; t += 8) {
            int hh = t >> 4, mt = (t >> 2) & 3, nt = t & 3;
            int h  = h0 + hh;
            v16bf a  = loadA(&QKs[0][h][mt * 16][0], DP, lane);
            v16bf bb = loadBT(&QKs[1][h][nt * 16][0], DP, lane);
            v8f acc = {};
            acc = __builtin_amdgcn_wmma_f32_16x16x32_bf16(false, a, false, bb,
                                                          (short)0, acc, false, false);
            int n0 = mt * 16 + ((lane >> 4) << 3);
            int m  = nt * 16 + (lane & 15);
            int i2 = m >> 3, j2 = m & 7;
            int pm = (edgeH ? (i2 < 4 ? 3 : 6) : 0) + (edgeW ? (j2 < 4 ? 1 : 2) : 0);
#pragma unroll
            for (int r = 0; r < 8; ++r) {
                int n  = n0 + r;
                int i1 = n >> 3, j1 = n & 7;
                int rel  = (i1 - i2 + 7) * 15 + (j1 - j2 + 7);
                float bv = BsL[rel * NH + h];
                int pn = (edgeH ? (i1 < 4 ? 3 : 6) : 0) + (edgeW ? (j1 < 4 ? 1 : 2) : 0);
                float msk = (pn != pm) ? -100.0f : 0.0f;
                Ss[hh][n][m] = acc[r] + bv + msk;
            }
        }
        __syncthreads();

        // softmax: 2 threads/row, all 256 threads; lane-pair shfl reduce (wave32)
        {
            int hh   = tid >> 7;
            int row  = (tid >> 1) & 63;
            int half = tid & 1;
            const float* srow = &Ss[hh][row][half * 32];
            float mx = -1e30f;
#pragma unroll
            for (int m = 0; m < 32; ++m) mx = fmaxf(mx, srow[m]);
            mx = fmaxf(mx, __shfl_xor(mx, 1));
            float s = 0.f;
            float ex[32];
#pragma unroll
            for (int m = 0; m < 32; ++m) { ex[m] = __expf(srow[m] - mx); s += ex[m]; }
            s += __shfl_xor(s, 1);
            float inv = __frcp_rn(s);
            uint16_t* prow = &Ps[hh][row][half * 32];
#pragma unroll
            for (int m = 0; m < 32; ++m) prow[m] = f2bf(ex[m] * inv);
        }
        __syncthreads();

        // O = P * V : 8 tiles (2 heads x 4 row-tiles), one per wave
        {
            int hh = wid >> 2, mt = wid & 3;
            int h  = h0 + hh;
            v8f acc = {};
#pragma unroll
            for (int kt = 0; kt < 2; ++kt) {
                v16bf a  = loadA(&Ps[hh][mt * 16][0] + kt * 32, NTOK, lane);
                v16bf bb = loadBT(&VTs[h][0][0] + kt * 32, NTOK, lane);
                acc = __builtin_amdgcn_wmma_f32_16x16x32_bf16(false, a, false, bb,
                                                              (short)0, acc, false, false);
            }
            int n0 = mt * 16 + ((lane >> 4) << 3);
            int dd = lane & 15;
            if (dd < DH) {
                uint16_t* dst = &Os[n0][h * DH + dd];
#pragma unroll
                for (int r = 0; r < 8; ++r)
                    dst[r * CP] = f2bf(acc[r]);
            }
        }
        __syncthreads();
    }

    // ======== projection: Y = Os[64 x 64] * Wp + b, reverse-shift store ========
    for (int t = wid; t < 16; t += 8) {
        int mt = t & 3, nt = t >> 2;
        v8f acc = {};
#pragma unroll
        for (int kt = 0; kt < 2; ++kt) {
            v16bf a  = loadA(&Os[mt * 16][0] + kt * 32, CP, lane);
            v16bf bb = loadBT(&WpT[nt * 16][0] + kt * 32, CP, lane);
            acc = __builtin_amdgcn_wmma_f32_16x16x32_bf16(false, a, false, bb,
                                                          (short)0, acc, false, false);
        }
        int n0 = mt * 16 + ((lane >> 4) << 3);
        int c  = nt * 16 + (lane & 15);
        if (c < CCH) {
            float pb = proj_b[c];
#pragma unroll
            for (int r = 0; r < 8; ++r) {
                int n  = n0 + r;
                int ii = n >> 3, jj = n & 7;
                int ho = (wh * WSZ + ii + WSZ / 2) % HIMG;
                int wo = (ww * WSZ + jj + WSZ / 2) % WIMG;
                out[(((size_t)b * HIMG + ho) * WIMG + wo) * CCH + c] = acc[r] + pb;
            }
        }
    }
}

extern "C" void kernel_launch(void* const* d_in, const int* in_sizes, int n_in,
                              void* d_out, int out_size, void* d_ws, size_t ws_size,
                              hipStream_t stream) {
    (void)in_sizes; (void)n_in; (void)out_size; (void)ws_size;
    const float* img    = (const float*)d_in[0];
    const float* qkv_w  = (const float*)d_in[1];
    const float* bias_m = (const float*)d_in[2];
    const float* proj_w = (const float*)d_in[3];
    const float* proj_b = (const float*)d_in[4];
    float*    out  = (float*)d_out;
    uint32_t* blob = (uint32_t*)d_ws;

    swin_prep<<<1, 256, 0, stream>>>(qkv_w, proj_w, bias_m, blob);
    swin_wmsa_fused<<<BIMG * WNUM, 256, 0, stream>>>(img, blob, proj_b, out);
}
